// GCN_274877907322
// MI455X (gfx1250) — matrix operand order repaired and checked by reading, not compiled, not run on hardware
//
#include <hip/hip_runtime.h>

// ---------- vector types ----------
typedef __attribute__((ext_vector_type(16))) __bf16       v16bf;
typedef __attribute__((ext_vector_type(8)))  float        v8f;
typedef __attribute__((ext_vector_type(4)))  unsigned int u32x4;
typedef __attribute__((ext_vector_type(4)))  float        f32x4;

union Frag { u32x4 q[2]; v16bf v; };

__device__ __forceinline__ unsigned int f2bf(float f) {
  // round-to-nearest-even f32 -> bf16 (low 16 bits)
  unsigned int u = __float_as_uint(f);
  u += 0x7FFFu + ((u >> 16) & 1u);
  return u >> 16;
}
__device__ __forceinline__ unsigned int pk2(float a, float b) {
  return f2bf(a) | (f2bf(b) << 16);
}

// guaranteed hardware f32 atomic add (device scope)
__device__ __forceinline__ void atomic_add_f32(float* p, float v) {
#if __has_builtin(__builtin_amdgcn_global_atomic_fadd_f32)
  __builtin_amdgcn_global_atomic_fadd_f32(p, v);
#else
  asm volatile("global_atomic_add_f32 %0, %1, off scope:SCOPE_DEV"
               :: "v"(p), "v"(v) : "memory");
#endif
}

// ---------------------------------------------------------------------------
// GEMM:  Out[M,N] = (relu?)(A[M,K]) @ W[K,N] + bias[N]
// 256 threads (8 waves), block tile 64(M) x 64(N), BK=32.
// wave w: wm=w&3 -> 16-row stripe; wn=w>>2 -> 32-col stripe as two 16x16
// WMMA tiles (A fragment reused). K, N compile-time for clean addressing.
// ---------------------------------------------------------------------------
template<bool RELU_A, int K, int N>
__global__ __launch_bounds__(256)
void gemm_bias_wmma(const float* __restrict__ A, const float* __restrict__ W,
                    const float* __restrict__ bias, float* __restrict__ Out,
                    int M) {
  // padded row stride of 5 uint4 (20 dwords) -> conflict-free b128 LDS access
  __shared__ u32x4 AsQ[64 * 5];   // [row m][k 0..31] bf16
  __shared__ u32x4 BsQ[64 * 5];   // [col n][k 0..31] bf16 (W transposed)

  const int tid  = threadIdx.x;
  const int lane = tid & 31;
  const int wv   = tid >> 5;
  const int half = lane >> 4;
  const int l16  = lane & 15;
  const int wm   = wv & 3;
  const int wn   = wv >> 2;
  const int m0   = blockIdx.y * 64;
  const int n0   = blockIdx.x * 64;

  // A loader: (row ar, 8 cols at ac*8)
  const int  ar   = tid >> 2;
  const int  ac   = tid & 3;
  const bool a_ok = (m0 + ar) < M;
  const float* aptr = A + (size_t)(m0 + ar) * K + ac * 8;   // + k0 per iter
  // B loader: (col bj, 8 k's at bc8*8), coalesced in n
  const int  bj   = tid & 63;
  const int  bc8  = tid >> 6;
  const int  bn   = n0 + bj;
  const bool b_ok = bn < N;
  const float* wptr = W + (size_t)(bc8 * 8) * N + bn;       // + k0*N per iter

  v8f acc0 = {0.f,0.f,0.f,0.f,0.f,0.f,0.f,0.f};
  v8f acc1 = {0.f,0.f,0.f,0.f,0.f,0.f,0.f,0.f};

  for (int k0 = 0; k0 < K; k0 += 32) {
    // ---- global -> registers ----
    f32x4 alo = {0.f,0.f,0.f,0.f}, ahi = {0.f,0.f,0.f,0.f};
    if (a_ok) {
      const f32x4* p = (const f32x4*)(aptr + k0);
      alo = p[0]; ahi = p[1];
      if (RELU_A) {
        alo.x = fmaxf(alo.x, 0.f); alo.y = fmaxf(alo.y, 0.f);
        alo.z = fmaxf(alo.z, 0.f); alo.w = fmaxf(alo.w, 0.f);
        ahi.x = fmaxf(ahi.x, 0.f); ahi.y = fmaxf(ahi.y, 0.f);
        ahi.z = fmaxf(ahi.z, 0.f); ahi.w = fmaxf(ahi.w, 0.f);
      }
    }
    float wv8[8] = {0.f,0.f,0.f,0.f,0.f,0.f,0.f,0.f};
    if (b_ok) {
      const float* p = wptr + (size_t)k0 * N;   // N constexpr -> imm offsets
#pragma unroll
      for (int i = 0; i < 8; ++i) wv8[i] = p[i * N];
    }

    __syncthreads();   // previous iteration's fragment reads done
    u32x4 pa = { pk2(alo.x, alo.y), pk2(alo.z, alo.w),
                 pk2(ahi.x, ahi.y), pk2(ahi.z, ahi.w) };
    AsQ[ar * 5 + ac] = pa;
    u32x4 pb = { pk2(wv8[0], wv8[1]), pk2(wv8[2], wv8[3]),
                 pk2(wv8[4], wv8[5]), pk2(wv8[6], wv8[7]) };
    BsQ[bj * 5 + bc8] = pb;
    __syncthreads();

    // ---- fragments (ISA VGPR layouts) ----
    // A 16x32: half0 -> K{0..7}@u4[0], K{16..23}@u4[2]; half1 -> +8 / +8
    Frag fa;
    const int mrow = wm * 16 + l16;
    fa.q[0] = AsQ[mrow * 5 + half];
    fa.q[1] = AsQ[mrow * 5 + 2 + half];
    // B 32x16: half0 -> K{0..7}@u4[0], K{16..23}@u4[2]; half1 -> +8 / +8
    Frag fb0, fb1;
    const int nrow0 = wn * 32 + l16;
    fb0.q[0] = BsQ[nrow0 * 5 + half];
    fb0.q[1] = BsQ[nrow0 * 5 + 2 + half];
    const int nrow1 = nrow0 + 16;
    fb1.q[0] = BsQ[nrow1 * 5 + half];
    fb1.q[1] = BsQ[nrow1 * 5 + 2 + half];

    acc0 = __builtin_amdgcn_wmma_f32_16x16x32_bf16(
        false, fa.v, false, fb0.v, (short)0, acc0, false, false);
    acc1 = __builtin_amdgcn_wmma_f32_16x16x32_bf16(
        false, fa.v, false, fb1.v, (short)0, acc1, false, false);
  }

  // ---- store D + bias ----  (C/D: VGPR r -> M = r + 8*half, N = l16)
#pragma unroll
  for (int tn = 0; tn < 2; ++tn) {
    const v8f acc = tn ? acc1 : acc0;
    const int n = n0 + wn * 32 + tn * 16 + l16;
    if (n < N) {
      const float bv = bias[n];
#pragma unroll
      for (int r = 0; r < 8; ++r) {
        const int m = m0 + wm * 16 + 8 * half + r;
        if (m < M) Out[(size_t)m * N + n] = acc[r] + bv;
      }
    }
  }
}

// ---------------------------------------------------------------------------
// SpMM scatter:  H[dst,:] += val * S[src,:]  (thread = edge x 4-float chunk)
// ---------------------------------------------------------------------------
template<int CPE>   // chunks of 4 floats per edge; feature dim d = 4*CPE
__global__ __launch_bounds__(256)
void spmm_scatter(const int* __restrict__ src, const int* __restrict__ dst,
                  const float* __restrict__ val, const float* __restrict__ S,
                  float* __restrict__ H, unsigned total) {
  const unsigned gid = blockIdx.x * 256u + threadIdx.x;
  if (gid >= total) return;
  const unsigned e = gid / (unsigned)CPE;   // constant divisor
  const unsigned c = gid % (unsigned)CPE;
  const int d = CPE * 4;
  const int   sn = src[e];
  const int   tn = dst[e];
  const float w  = val[e];
  const f32x4 v  = ((const f32x4*)(S + (size_t)sn * d))[c];
  float* hp = H + (size_t)tn * d + c * 4u;
  atomic_add_f32(hp + 0, w * v.x);
  atomic_add_f32(hp + 1, w * v.y);
  atomic_add_f32(hp + 2, w * v.z);
  atomic_add_f32(hp + 3, w * v.w);
}

__global__ __launch_bounds__(256)
void zero_f32x4(f32x4* __restrict__ p, unsigned n4) {
  const unsigned i = blockIdx.x * 256u + threadIdx.x;
  if (i < n4) p[i] = (f32x4){0.f, 0.f, 0.f, 0.f};
}

// ---------------------------------------------------------------------------
// In-place row-wise log_softmax over 40 classes; one wave32 per row.
// ---------------------------------------------------------------------------
__global__ __launch_bounds__(256)
void log_softmax40(float* __restrict__ out, int M) {
  const int wid  = (int)((blockIdx.x * 256u + threadIdx.x) >> 5);  // uniform/wave
  const int lane = threadIdx.x & 31;
  if (wid >= M) return;
  float* row = out + (size_t)wid * 40;
  const float v0 = row[lane];
  const float v1 = (lane < 8) ? row[32 + lane] : -3.4e38f;
  float m = fmaxf(v0, v1);
#pragma unroll
  for (int off = 16; off; off >>= 1) m = fmaxf(m, __shfl_xor(m, off, 32));
  float s = __expf(v0 - m) + ((lane < 8) ? __expf(v1 - m) : 0.f);
#pragma unroll
  for (int off = 16; off; off >>= 1) s += __shfl_xor(s, off, 32);
  const float lse = m + __logf(s);
  row[lane] = v0 - lse;
  if (lane < 8) row[32 + lane] = v1 - lse;
}

// ---------------------------------------------------------------------------
extern "C" void kernel_launch(void* const* d_in, const int* in_sizes, int n_in,
                              void* d_out, int out_size, void* d_ws, size_t ws_size,
                              hipStream_t stream) {
  const float* x    = (const float*)d_in[0];
  const int*   esrc = (const int*)  d_in[1];
  const int*   edst = (const int*)  d_in[2];
  const float* ev   = (const float*)d_in[3];
  const float* W0   = (const float*)d_in[4];
  const float* b0   = (const float*)d_in[5];
  const float* W1   = (const float*)d_in[6];
  const float* b1   = (const float*)d_in[7];
  const float* W2   = (const float*)d_in[8];
  const float* b2   = (const float*)d_in[9];

  const int      M = 100000;
  const unsigned E = (unsigned)in_sizes[1];

  float* S = (float*)d_ws;                                             // <=102.4 MB
  float* H = (float*)((char*)d_ws + (size_t)M * 256 * sizeof(float));  // <=102.4 MB
  float* out = (float*)d_out;

  const dim3 blk(256);
  const int  mb = (M + 63) / 64;

  // ---- layer 0: S = x@W0+b0 ; H = scatter(S)  (relu fused into next A-load)
  gemm_bias_wmma<false, 512, 256><<<dim3(4, mb), blk, 0, stream>>>(x, W0, b0, S, M);
  {
    const unsigned n4 = (unsigned)M * 64u;
    zero_f32x4<<<(n4 + 255) / 256, blk, 0, stream>>>((f32x4*)H, n4);
    const unsigned total = E * 64u;
    spmm_scatter<64><<<(total + 255) / 256, blk, 0, stream>>>(esrc, edst, ev, S, H, total);
  }
  // ---- layer 1: S = relu(H)@W1+b1 ; H = scatter(S)
  gemm_bias_wmma<true, 256, 128><<<dim3(2, mb), blk, 0, stream>>>(H, W1, b1, S, M);
  {
    const unsigned n4 = (unsigned)M * 32u;
    zero_f32x4<<<(n4 + 255) / 256, blk, 0, stream>>>((f32x4*)H, n4);
    const unsigned total = E * 32u;
    spmm_scatter<32><<<(total + 255) / 256, blk, 0, stream>>>(esrc, edst, ev, S, H, total);
  }
  // ---- layer 2: S = relu(H)@W2+b2 ; out = scatter(S) ; log_softmax in place
  gemm_bias_wmma<true, 128, 40><<<dim3(1, mb), blk, 0, stream>>>(H, W2, b2, S, M);
  {
    const unsigned n4 = (unsigned)M * 10u;
    zero_f32x4<<<(n4 + 255) / 256, blk, 0, stream>>>((f32x4*)out, n4);
    const unsigned total = E * 10u;
    spmm_scatter<10><<<(total + 255) / 256, blk, 0, stream>>>(esrc, edst, ev, S, out, total);
  }
  log_softmax40<<<(M + 7) / 8, blk, 0, stream>>>(out, M);
}
